// PredictorLG_27487790694698
// MI455X (gfx1250) — matrix operand order repaired and checked by reading, not compiled, not run on hardware
//
#include <hip/hip_runtime.h>

typedef __bf16 bf16_t;
typedef __attribute__((ext_vector_type(16))) __bf16 v16bf;
typedef __attribute__((ext_vector_type(8)))  __bf16 bf16x8;
typedef __attribute__((ext_vector_type(4)))  __bf16 bf16x4;
typedef __attribute__((ext_vector_type(8)))  float  v8f;
typedef __attribute__((ext_vector_type(4)))  float  f32x4;

#define MMOD 2
#define CDIM 384
#define BN   131072          // B*N rows per modality
#define ROWS_PER_WG 128      // 8 row-tiles of 16
#define WAVES 8

// packed-fragment counts per modality (fragment = 512 bf16 = one 32x16 B tile)
#define FRAGS1 (12*24)       // 384x384 : KT=12, NT=24
#define FRAGS2 (12*12)       // 384x192 : KT=12, NT=12
#define FRAGS3 (6*6)         // 192x96  : KT=6,  NT=6
#define FRAGS4 (3*1)         // 96x2    : KT=3,  NT=1 (zero padded to N=16)
#define OFF1 0
#define OFF2 (OFF1 + MMOD*FRAGS1*512)
#define OFF3 (OFF2 + MMOD*FRAGS2*512)
#define OFF4 (OFF3 + MMOD*FRAGS3*512)

// ---------------------------------------------------------------------------
// Repack fp32 weights [m, K, N] into bf16 WMMA B-fragments.
// Fragment element e: lane = e>>4, i = e&15 ; K = kt*32 + (i&7) + 16*(i>>3) + 8*(lane>=16),
// N = nt*16 + (lane&15)  (per CDNA5 ISA 16-bit 32x16 B layout).
// ---------------------------------------------------------------------------
__global__ void pack_wmma_b(const float* __restrict__ W, bf16_t* __restrict__ dst,
                            int K, int N, int KT, int NT)
{
    const int fpm = KT * NT;
    const int f   = blockIdx.x % fpm;
    const int m   = blockIdx.x / fpm;
    const int nt  = f / KT;
    const int kt  = f % KT;
    const float* Wm = W + (size_t)m * K * N;
    bf16_t* D = dst + ((size_t)m * fpm + f) * 512;
    for (int e = threadIdx.x; e < 512; e += blockDim.x) {
        const int lane = e >> 4;
        const int i    = e & 15;
        const int k = kt * 32 + (i & 7) + ((i >> 3) * 16) + ((lane >= 16) ? 8 : 0);
        const int n = nt * 16 + (lane & 15);
        const float v = (k < K && n < N) ? Wm[(size_t)k * N + n] : 0.0f;
        D[e] = (bf16_t)v;
    }
}

// ---------------------------------------------------------------------------
// Load a 16x32 bf16 A fragment from row-major LDS tile (two ds_load_b128).
// Lane m = lane&15; halves K = {off..off+7, off+16..off+23}, off = 8*(lane>=16)
// ---------------------------------------------------------------------------
__device__ __forceinline__ v16bf load_a_frag(const bf16_t* A, int lda, int row,
                                             int kstep, int lane)
{
    const bf16_t* p = A + row * lda + kstep * 32 + ((lane & 16) ? 8 : 0);
    bf16x8 lo = *(const bf16x8*)p;
    bf16x8 hi = *(const bf16x8*)(p + 16);
    return __builtin_shufflevector(lo, hi, 0,1,2,3,4,5,6,7,8,9,10,11,12,13,14,15);
}

// Branch-free tanh-form GELU: 1 trans op (v_exp_f32), no libm call, no branches.
__device__ __forceinline__ float gelu_fast(float v)
{
    const float u  = 0.7978845608028654f * v * (1.0f + 0.044715f * v * v);
    const float t  = __expf(-2.0f * fabsf(u));          // in (0,1]
    const float th = __builtin_copysignf(__fdividef(1.0f - t, 1.0f + t), u);
    return 0.5f * v * (1.0f + th);
}

// ---------------------------------------------------------------------------
// n-tile-split MLP layer: wave w owns n-tiles {w, w+8, ...} and computes them
// for all 8 row-tiles with 8 independent accumulators. Each weight fragment
// is loaded from L2 exactly once per workgroup; 8 independent WMMAs pipeline
// per fragment (no serialized accumulation chain).
// ---------------------------------------------------------------------------
template<int KT, int NT, int LDA, int LDO>
__device__ __forceinline__ void mlp_layer_nt(const bf16_t* __restrict__ A,
                                             const bf16_t* __restrict__ Wp,
                                             const float*  __restrict__ bias,
                                             bf16_t* __restrict__ O,
                                             int wave, int lane)
{
    const int col = lane & 15;
    const int rhi = (lane & 16) ? 8 : 0;    // C/D layout: lanes 16-31 hold M=8..15
    for (int nt = wave; nt < NT; nt += WAVES) {
        const bf16_t* wfrag = Wp + (size_t)nt * KT * 512 + lane * 16;
        const float b = bias[nt * 16 + col];
        v8f acc[8];
        #pragma unroll
        for (int rt = 0; rt < 8; ++rt) {
            #pragma unroll
            for (int i = 0; i < 8; ++i) acc[rt][i] = b;
        }
        for (int kt = 0; kt < KT; ++kt) {
            v16bf bw = *(const v16bf*)(wfrag + (size_t)kt * 512);
            // high-locality -> WGP-scope prefetch of next fragment into near cache
            __builtin_prefetch(wfrag + (size_t)(kt + 1) * 512, 0, 3);
            #pragma unroll
            for (int rt = 0; rt < 8; ++rt) {
                v16bf a = load_a_frag(A, LDA, rt * 16 + col, kt, lane);
                acc[rt] = __builtin_amdgcn_wmma_f32_16x16x32_bf16(
                              false, a, false, bw, (short)0, acc[rt], false, false);
            }
        }
        #pragma unroll
        for (int rt = 0; rt < 8; ++rt) {
            #pragma unroll
            for (int r = 0; r < 8; ++r) {
                const float v = gelu_fast(acc[rt][r]);
                O[(rt * 16 + r + rhi) * LDO + nt * 16 + col] = (bf16_t)v;
            }
        }
    }
}

// ---------------------------------------------------------------------------
// Fully fused: LayerNorm -> 4x GEMM(+GELU) -> log_softmax, one pass over x.
// ---------------------------------------------------------------------------
__global__ __launch_bounds__(256) void predictor_fused(
    const float* __restrict__ x,
    const float* __restrict__ ln_g, const float* __restrict__ ln_b,
    const float* __restrict__ b1, const float* __restrict__ b2,
    const float* __restrict__ b3, const float* __restrict__ b4,
    const bf16_t* __restrict__ wp1, const bf16_t* __restrict__ wp2,
    const bf16_t* __restrict__ wp3, const bf16_t* __restrict__ wp4,
    float* __restrict__ out)
{
    extern __shared__ bf16_t smem[];
    bf16_t* A0 = smem;                          // 128 x 384 bf16
    bf16_t* A1 = smem + ROWS_PER_WG * CDIM;     // 128 x 384 bf16
    bf16_t* A2 = smem;                          // 128 x 192 (aliases A0, after barrier)
    bf16_t* A3 = smem + ROWS_PER_WG * 192;      // 128 x  96 (inside A0 buffer)

    const int m    = blockIdx.x >> 10;          // BN/128 = 1024 tiles per modality
    const int tile = blockIdx.x & 1023;
    const int wave = threadIdx.x >> 5;
    const int lane = threadIdx.x & 31;
    const size_t rowBase = (size_t)tile * ROWS_PER_WG;

    // ---- Stage 1: LayerNorm (fp32) -> bf16 rows in LDS (row-split) ----
    {
        const int c0 = lane * 4;
        const f32x4* gv = (const f32x4*)(ln_g + m * CDIM);
        const f32x4* bv = (const f32x4*)(ln_b + m * CDIM);
        f32x4 g0 = gv[lane], g1 = gv[lane + 32], g2 = gv[lane + 64];
        f32x4 e0 = bv[lane], e1 = bv[lane + 32], e2 = bv[lane + 64];
        for (int r = 0; r < 16; ++r) {
            const size_t grow = (size_t)m * BN + rowBase + wave * 16 + r;
            const f32x4* xr = (const f32x4*)(x + grow * CDIM);
            f32x4 v0 = xr[lane], v1 = xr[lane + 32], v2 = xr[lane + 64];
            float s = 0.f, ss = 0.f;
            #pragma unroll
            for (int j = 0; j < 4; ++j) {
                s  += v0[j] + v1[j] + v2[j];
                ss += v0[j]*v0[j] + v1[j]*v1[j] + v2[j]*v2[j];
            }
            #pragma unroll
            for (int o = 16; o > 0; o >>= 1) {
                s  += __shfl_xor(s,  o, 32);
                ss += __shfl_xor(ss, o, 32);
            }
            const float mu  = s  * (1.0f / CDIM);
            const float var = ss * (1.0f / CDIM) - mu * mu;
            const float inv = rsqrtf(var + 1e-5f);
            bf16_t* orow = A0 + (wave * 16 + r) * CDIM;
            bf16x4 o0, o1, o2;
            #pragma unroll
            for (int j = 0; j < 4; ++j) {
                o0[j] = (bf16_t)((v0[j] - mu) * inv * g0[j] + e0[j]);
                o1[j] = (bf16_t)((v1[j] - mu) * inv * g1[j] + e1[j]);
                o2[j] = (bf16_t)((v2[j] - mu) * inv * g2[j] + e2[j]);
            }
            *(bf16x4*)(orow + c0)       = o0;
            *(bf16x4*)(orow + c0 + 128) = o1;
            *(bf16x4*)(orow + c0 + 256) = o2;
        }
    }
    __syncthreads();

    // ---- Stages 2-4: n-tile-split WMMA GEMM + bias + GELU ----
    mlp_layer_nt<12, 24, CDIM, CDIM>(A0, wp1 + (size_t)m * FRAGS1 * 512,
                                     b1 + m * CDIM, A1, wave, lane);
    __syncthreads();
    mlp_layer_nt<12, 12, CDIM, 192>(A1, wp2 + (size_t)m * FRAGS2 * 512,
                                    b2 + m * 192, A2, wave, lane);
    __syncthreads();
    mlp_layer_nt<6, 6, 192, 96>(A2, wp3 + (size_t)m * FRAGS3 * 512,
                                b3 + m * 96, A3, wave, lane);
    __syncthreads();

    // ---- Stage 5: 96 -> 2 GEMM (row-split, N padded to 16) + log_softmax ----
    {
        const bf16_t* w4m = wp4 + (size_t)m * FRAGS4 * 512;
        const int col  = lane & 15;
        const int rhi  = (lane & 16) ? 8 : 0;
        const int arow = wave * 16 + col;
        const float b = (col < 2) ? b4[m * 2 + col] : 0.0f;
        v8f acc;
        #pragma unroll
        for (int i = 0; i < 8; ++i) acc[i] = b;
        #pragma unroll
        for (int kt = 0; kt < 3; ++kt) {
            v16bf a  = load_a_frag(A3, 96, arow, kt, lane);
            v16bf bw = *(const v16bf*)(w4m + kt * 512 + lane * 16);
            acc = __builtin_amdgcn_wmma_f32_16x16x32_bf16(
                      false, a, false, bw, (short)0, acc, false, false);
        }
        #pragma unroll
        for (int r = 0; r < 8; ++r) {
            const float xv = acc[r];
            const float yv = __shfl_xor(xv, 1, 32);   // pair n=0 <-> n=1
            const float mx = fmaxf(xv, yv);
            const float lse = mx + __logf(__expf(xv - mx) + __expf(yv - mx));
            if (col < 2) {
                const size_t grow = (size_t)m * BN + rowBase + wave * 16 + r + rhi;
                out[grow * 2 + col] = xv - lse;
            }
        }
    }
}

extern "C" void kernel_launch(void* const* d_in, const int* in_sizes, int n_in,
                              void* d_out, int out_size, void* d_ws, size_t ws_size,
                              hipStream_t stream)
{
    const float* x   = (const float*)d_in[0];
    const float* lng = (const float*)d_in[1];
    const float* lnb = (const float*)d_in[2];
    const float* W1  = (const float*)d_in[3];
    const float* b1  = (const float*)d_in[4];
    const float* W2  = (const float*)d_in[5];
    const float* b2  = (const float*)d_in[6];
    const float* W3  = (const float*)d_in[7];
    const float* b3  = (const float*)d_in[8];
    const float* W4  = (const float*)d_in[9];
    const float* b4  = (const float*)d_in[10];
    float*  out = (float*)d_out;
    bf16_t* ws  = (bf16_t*)d_ws;   // needs ~965 KB for packed bf16 weights

    pack_wmma_b<<<MMOD * FRAGS1, 256, 0, stream>>>(W1, ws + OFF1, 384, 384, 12, 24);
    pack_wmma_b<<<MMOD * FRAGS2, 256, 0, stream>>>(W2, ws + OFF2, 384, 192, 12, 12);
    pack_wmma_b<<<MMOD * FRAGS3, 256, 0, stream>>>(W3, ws + OFF3, 192,  96,  6,  6);
    pack_wmma_b<<<MMOD * FRAGS4, 256, 0, stream>>>(W4, ws + OFF4,  96,   2,  3,  1);

    const size_t shmem = (size_t)2 * ROWS_PER_WG * CDIM * sizeof(bf16_t); // 192 KB
    predictor_fused<<<MMOD * (BN / ROWS_PER_WG), 256, shmem, stream>>>(
        x, lng, lnb, b1, b2, b3, b4,
        ws + OFF1, ws + OFF2, ws + OFF3, ws + OFF4, out);
}
